// TtShiftedWindowAttention_25409026523919
// MI455X (gfx1250) — matrix operand order repaired and checked by reading, not compiled, not run on hardware
//
#include <hip/hip_runtime.h>
#include <hip/hip_bf16.h>

typedef __attribute__((ext_vector_type(16))) _Float16 v16h;
typedef __attribute__((ext_vector_type(8)))  float    v8f;

#define N_TOK 49
#define HD    32

// region id for the shift mask: rows/cols [0,49) | [49,53) | [53,56)
__device__ __forceinline__ int regf(int r) { return r < 49 ? 0 : (r < 53 ? 1 : 2); }

// LDS byte offset from a generic pointer (LDS aperture: addr[31:0] = LDS offset)
__device__ __forceinline__ unsigned lds_off(const void* p) {
    return (unsigned)(unsigned long long)p;
}
// async 16B copy global -> LDS (ASYNCcnt tracked)
__device__ __forceinline__ void async_cp16(unsigned lds_addr, const void* gptr) {
    asm volatile("global_load_async_to_lds_b128 %0, %1, off"
                 :: "v"(lds_addr), "v"((unsigned long long)gptr) : "memory");
}
__device__ __forceinline__ void async_wait0() {
    asm volatile("s_wait_asynccnt 0x0" ::: "memory");
}

// ---------------------------------------------------------------------------
// Kernel 0: one-shot weight conversion f32 -> f16
// ---------------------------------------------------------------------------
__global__ __launch_bounds__(256) void cvt_weights(
    const float* __restrict__ qkv_w, const float* __restrict__ proj_w,
    _Float16* __restrict__ qw_h, _Float16* __restrict__ pw_h)
{
    int i = blockIdx.x * 256 + threadIdx.x;
    if (i < 196608) qw_h[i] = (_Float16)qkv_w[i];   // 768 x 256
    if (i < 65536)  pw_h[i] = (_Float16)proj_w[i];  // 256 x 256
}

// ---------------------------------------------------------------------------
// Kernel 1: fused shift + window-partition + QKV GEMM (M=64(49), K=256, N=768)
// One block per window (2048 blocks, 256 threads = 8 waves).
// qkv_out layout: [w][s(qkv)][head][tok 49][ch 32]  (f16), q pre-scaled.
// ---------------------------------------------------------------------------
__global__ __launch_bounds__(256) void qkv_kernel(
    const float* __restrict__ x, const _Float16* __restrict__ qw_h,
    const float* __restrict__ qkv_b, _Float16* __restrict__ qkv_out)
{
    __shared__ __align__(16) _Float16 At[64 * 256];

    const int w  = blockIdx.x;
    const int b  = w >> 6;
    const int wi = (w >> 3) & 7;
    const int wj = w & 7;
    const int tid = threadIdx.x;

    // stage shifted window tile into LDS as f16 (rows 49..63 zero)
    for (int m = 0; m < 64; ++m) {
        _Float16 h = (_Float16)0.f;
        if (m < N_TOK) {
            int tr = m / 7, tc = m - (m / 7) * 7;
            int sr = wi * 7 + tr + 3; if (sr >= 56) sr -= 56;   // roll(-3)
            int sc = wj * 7 + tc + 3; if (sc >= 56) sc -= 56;
            const float* src = &x[(((size_t)b * 56 + sr) * 56 + sc) * 256 + tid];
            __builtin_prefetch(src + 4 * 56 * 256, 0, 0);       // global_prefetch_b8
            h = (_Float16)(*src);
        }
        At[m * 256 + tid] = h;
    }
    __syncthreads();

    const int lane = tid & 31;
    const int wv   = tid >> 5;
    const int lo   = lane & 15;
    const int hi   = lane >> 4;
    const float qscale = 0.17677669529663687f;  // 1/sqrt(32)

    for (int i = 0; i < 6; ++i) {               // 6 N-tiles per wave (48 total)
        const int o = (wv * 6 + i) * 16 + lo;   // output feature for this lane
        v8f acc[4] = {};
        for (int kk = 0; kk < 8; ++kk) {
            const int kb = kk * 32;
            // B fragment: 16 contiguous f16 from pre-converted weights
            v16h bf;
            const uint4* wp = (const uint4*)&qw_h[(size_t)o * 256 + kb + hi * 16];
            *(uint4*)&bf       = wp[0];
            *((uint4*)&bf + 1) = wp[1];
#pragma unroll
            for (int mt = 0; mt < 4; ++mt) {
                const int m = mt * 16 + lo;
                v16h af;
                *(uint4*)&af       = *(const uint4*)&At[m * 256 + kb + hi * 16];
                *((uint4*)&af + 1) = *(const uint4*)&At[m * 256 + kb + hi * 16 + 8];
                acc[mt] = __builtin_amdgcn_wmma_f32_16x16x32_f16(
                    false, af, false, bf, (short)0, acc[mt], false, false);
            }
        }
        // epilogue: +bias, scale q, store f16
        const int s    = o >> 8;
        const int head = (o >> 5) & 7;
        const int ch   = o & 31;
        const float bias  = qkv_b[o];
        const float scale = (s == 0) ? qscale : 1.0f;
        const size_t base = (((size_t)w * 3 + s) * 8 + head) * (size_t)(N_TOK * HD) + ch;
#pragma unroll
        for (int mt = 0; mt < 4; ++mt)
#pragma unroll
            for (int r = 0; r < 8; ++r) {
                const int m = mt * 16 + r + hi * 8;
                if (m < N_TOK)
                    qkv_out[base + (size_t)m * HD] = (_Float16)((acc[mt][r] + bias) * scale);
            }
    }
}

// ---------------------------------------------------------------------------
// Kernel 2: per (window, head) attention: logits + rel-bias + shift-mask +
// softmax + attn@v.  One block per (w,head): 16384 blocks, 128 threads = 4 waves.
// ---------------------------------------------------------------------------
__global__ __launch_bounds__(128) void attn_kernel(
    const _Float16* __restrict__ qkv, const float* __restrict__ bias_table,
    _Float16* __restrict__ attn_out)
{
    __shared__ __align__(16) _Float16 qb[64 * 32];
    __shared__ __align__(16) _Float16 kb[64 * 32];
    __shared__ __align__(16) _Float16 vb[64 * 32];
    __shared__ __align__(16) _Float16 pb[64 * 64];

    const int w    = blockIdx.x >> 3;
    const int head = blockIdx.x & 7;
    const int wi   = (w >> 3) & 7;
    const int wj   = w & 7;
    const int tid  = threadIdx.x;

    // async-stage q,k,v (49x32 f16 = 196 x 16B chunks each); rows 49..63 zeroed
    const _Float16* gq = &qkv[(((size_t)w * 3 + 0) * 8 + head) * (N_TOK * HD)];
    const _Float16* gk = &qkv[(((size_t)w * 3 + 1) * 8 + head) * (N_TOK * HD)];
    const _Float16* gv = &qkv[(((size_t)w * 3 + 2) * 8 + head) * (N_TOK * HD)];
    for (int c = tid; c < 196; c += 128) {
        async_cp16(lds_off(&qb[c * 8]), gq + c * 8);
        async_cp16(lds_off(&kb[c * 8]), gk + c * 8);
        async_cp16(lds_off(&vb[c * 8]), gv + c * 8);
    }
    {   // zero pad rows 49..63
        unsigned int* q32 = (unsigned int*)qb;
        unsigned int* k32 = (unsigned int*)kb;
        unsigned int* v32 = (unsigned int*)vb;
        for (int idx = 784 + tid; idx < 1024; idx += 128) {
            q32[idx] = 0u; k32[idx] = 0u; v32[idx] = 0u;
        }
    }
    async_wait0();
    __syncthreads();

    const int lane = tid & 31;
    const int wv   = tid >> 5;       // wave owns rows 16*wv .. 16*wv+15
    const int lo   = lane & 15;
    const int hi   = lane >> 4;

    // ---- logits = q @ k^T ----
    v16h aq;
    {
        const int m = wv * 16 + lo;
        *(uint4*)&aq       = *(const uint4*)&qb[m * 32 + hi * 16];
        *((uint4*)&aq + 1) = *(const uint4*)&qb[m * 32 + hi * 16 + 8];
    }
    v8f lg[4];
#pragma unroll
    for (int nt = 0; nt < 4; ++nt) {
        const int mp = nt * 16 + lo;
        v16h bk;
        *(uint4*)&bk       = *(const uint4*)&kb[mp * 32 + hi * 16];
        *((uint4*)&bk + 1) = *(const uint4*)&kb[mp * 32 + hi * 16 + 8];
        v8f z = {};
        lg[nt] = __builtin_amdgcn_wmma_f32_16x16x32_f16(
            false, aq, false, bk, (short)0, z, false, false);
    }

    // ---- + rel bias + shift mask, softmax over m, write p (f16) to LDS ----
#pragma unroll
    for (int r = 0; r < 8; ++r) {
        const int n = wv * 16 + r + hi * 8;
        const int ti = n / 7, tj = n - (n / 7) * 7;
        const int labn = 3 * regf(wi * 7 + ti) + regf(wj * 7 + tj);
        float vals[4];
#pragma unroll
        for (int nt = 0; nt < 4; ++nt) {
            const int m = nt * 16 + lo;
            float v;
            if (m < N_TOK && n < N_TOK) {
                const int si = m / 7, sj = m - (m / 7) * 7;
                const int idx = (ti - si + 6) * 13 + (tj - sj + 6);
                const int labm = 3 * regf(wi * 7 + si) + regf(wj * 7 + sj);
                v = lg[nt][r] + bias_table[idx * 8 + head]
                  + ((labn != labm) ? -100.0f : 0.0f);
            } else if (m < N_TOK) {
                v = 0.0f;                    // dummy row n>=49
            } else {
                v = -1e30f;                  // padded column -> softmax 0
            }
            vals[nt] = v;
        }
        float mx = fmaxf(fmaxf(vals[0], vals[1]), fmaxf(vals[2], vals[3]));
        for (int off = 1; off < 16; off <<= 1) mx = fmaxf(mx, __shfl_xor(mx, off, 32));
        float sum = 0.f;
#pragma unroll
        for (int nt = 0; nt < 4; ++nt) { vals[nt] = __expf(vals[nt] - mx); sum += vals[nt]; }
        for (int off = 1; off < 16; off <<= 1) sum += __shfl_xor(sum, off, 32);
        const float inv = 1.0f / sum;
#pragma unroll
        for (int nt = 0; nt < 4; ++nt)
            pb[n * 64 + nt * 16 + lo] = (_Float16)(vals[nt] * inv);
    }
    __syncthreads();

    // ---- out = p @ v  (M=16 per wave, K=64, N=32) ----
    v8f outa[2] = {};
#pragma unroll
    for (int ks = 0; ks < 2; ++ks) {
        const int kbs = ks * 32;
        const int m = wv * 16 + lo;
        v16h ap;
        *(uint4*)&ap       = *(const uint4*)&pb[m * 64 + kbs + hi * 16];
        *((uint4*)&ap + 1) = *(const uint4*)&pb[m * 64 + kbs + hi * 16 + 8];
#pragma unroll
        for (int nt = 0; nt < 2; ++nt) {
            const int d = nt * 16 + lo;
            v16h bv;
#pragma unroll
            for (int j = 0; j < 16; ++j)
                bv[j] = vb[(kbs + hi * 16 + j) * 32 + d];   // rows>=49 are zero
            outa[nt] = __builtin_amdgcn_wmma_f32_16x16x32_f16(
                false, ap, false, bv, (short)0, outa[nt], false, false);
        }
    }
    // store f16 [tok][head*32+ch]
#pragma unroll
    for (int nt = 0; nt < 2; ++nt)
#pragma unroll
        for (int r = 0; r < 8; ++r) {
            const int n = wv * 16 + r + hi * 8;
            if (n < N_TOK)
                attn_out[((size_t)w * N_TOK + n) * 256 + head * 32 + nt * 16 + lo]
                    = (_Float16)outa[nt][r];
        }
}

// ---------------------------------------------------------------------------
// Kernel 3: proj GEMM (M=64(49), K=256, N=256) + window-reverse + roll(+3,+3).
// One block per window, 256 threads = 8 waves.
// ---------------------------------------------------------------------------
__global__ __launch_bounds__(256) void proj_kernel(
    const _Float16* __restrict__ ain, const _Float16* __restrict__ pw_h,
    const float* __restrict__ proj_b, float* __restrict__ out)
{
    __shared__ __align__(16) _Float16 At[64 * 256];

    const int w  = blockIdx.x;
    const int b  = w >> 6;
    const int wi = (w >> 3) & 7;
    const int wj = w & 7;
    const int tid = threadIdx.x;

    // async-stage the 49x256 f16 window slab (contiguous: 1568 x 16B chunks)
    const _Float16* src = &ain[(size_t)w * N_TOK * 256];
    for (int c = tid; c < 1568; c += 256)
        async_cp16(lds_off(&At[c * 8]), src + c * 8);
    {   // zero rows 49..63
        uint4 z; z.x = z.y = z.z = z.w = 0u;
        for (int c = 1568 + tid; c < 2048; c += 256) ((uint4*)At)[c] = z;
    }
    async_wait0();
    __syncthreads();

    const int lane = tid & 31;
    const int wv   = tid >> 5;
    const int lo   = lane & 15;
    const int hi   = lane >> 4;

    for (int i = 0; i < 2; ++i) {               // 2 N-tiles per wave (16 total)
        const int o = (wv * 2 + i) * 16 + lo;
        v8f acc[4] = {};
        for (int kk = 0; kk < 8; ++kk) {
            const int kb = kk * 32;
            v16h bf;
            const uint4* wp = (const uint4*)&pw_h[(size_t)o * 256 + kb + hi * 16];
            *(uint4*)&bf       = wp[0];
            *((uint4*)&bf + 1) = wp[1];
#pragma unroll
            for (int mt = 0; mt < 4; ++mt) {
                const int m = mt * 16 + lo;
                v16h af;
                *(uint4*)&af       = *(const uint4*)&At[m * 256 + kb + hi * 16];
                *((uint4*)&af + 1) = *(const uint4*)&At[m * 256 + kb + hi * 16 + 8];
                acc[mt] = __builtin_amdgcn_wmma_f32_16x16x32_f16(
                    false, af, false, bf, (short)0, acc[mt], false, false);
            }
        }
        const float pbv = proj_b[o];
#pragma unroll
        for (int mt = 0; mt < 4; ++mt)
#pragma unroll
            for (int r = 0; r < 8; ++r) {
                const int m = mt * 16 + r + hi * 8;
                if (m < N_TOK) {
                    const int tr = m / 7, tc = m - (m / 7) * 7;
                    int row = wi * 7 + tr + 3; if (row >= 56) row -= 56;  // roll(+3)
                    int col = wj * 7 + tc + 3; if (col >= 56) col -= 56;
                    out[(((size_t)b * 56 + row) * 56 + col) * 256 + o] = acc[mt][r] + pbv;
                }
            }
    }
}

extern "C" void kernel_launch(void* const* d_in, const int* in_sizes, int n_in,
                              void* d_out, int out_size, void* d_ws, size_t ws_size,
                              hipStream_t stream) {
    (void)in_sizes; (void)n_in; (void)out_size; (void)ws_size;
    const float* x        = (const float*)d_in[0];
    const float* qkv_w    = (const float*)d_in[1];
    const float* qkv_b    = (const float*)d_in[2];
    const float* proj_w   = (const float*)d_in[3];
    const float* proj_b   = (const float*)d_in[4];
    const float* rel_bias = (const float*)d_in[5];

    // workspace layout (f16): [qkv_w 196608][proj_w 65536][qkv 77070336][attn 25690112]
    _Float16* qw_h    = (_Float16*)d_ws;
    _Float16* pw_h    = qw_h + 196608;
    _Float16* qkv_ws  = pw_h + 65536;
    _Float16* attn_ws = qkv_ws + (size_t)2048 * 3 * 8 * N_TOK * HD;

    cvt_weights<<<768,      256, 0, stream>>>(qkv_w, proj_w, qw_h, pw_h);
    qkv_kernel <<<2048,     256, 0, stream>>>(x, qw_h, qkv_b, qkv_ws);
    attn_kernel<<<2048 * 8, 128, 0, stream>>>(qkv_ws, rel_bias, attn_ws);
    proj_kernel<<<2048,     256, 0, stream>>>(attn_ws, pw_h, proj_b, (float*)d_out);
}
